// Feature_71975061946535
// MI455X (gfx1250) — matrix operand order repaired and checked by reading, not compiled, not run on hardware
//
#include <hip/hip_runtime.h>
#include <hip/hip_bf16.h>

typedef __attribute__((ext_vector_type(16))) _Float16 v16h;
typedef __attribute__((ext_vector_type(8)))  _Float16 v8h;
typedef __attribute__((ext_vector_type(8)))  float    v8f;
typedef __attribute__((ext_vector_type(4)))  int      v4i;

#define BATCH 8
#define NATOM 384
#define NNBR  383
#define NPAD  448      // max padded neighbor rows (actual <= 432)
#define NKT   14       // NPAD / 32 K-tiles
#define NNC   7        // 112 / 16 output-column chunks
#define WFRAG 18       // 4 W1 frags + 14 W2 frags, 512 halfs each

#if __has_builtin(__builtin_amdgcn_global_load_async_to_lds_b128)
#define HAVE_ASYNC_LDS 1
#else
#define HAVE_ASYNC_LDS 0
#endif

__device__ __forceinline__ float fast_tanh(float x) {
#if __has_builtin(__builtin_amdgcn_tanhf)
  return __builtin_amdgcn_tanhf(x);          // gfx1250 v_tanh_f32
#else
  return tanhf(x);
#endif
}

__device__ __forceinline__ void wait_async0() {
#if __has_builtin(__builtin_amdgcn_s_wait_asynccnt)
  __builtin_amdgcn_s_wait_asynccnt(0);
#else
  asm volatile("s_wait_asynccnt 0" ::: "memory");
#endif
}

__device__ __forceinline__ void lds_wave_fence() {
  // same-wave LDS ops are in-order per ISA; this stops compiler reordering
  __builtin_amdgcn_wave_barrier();
  asm volatile("s_wait_dscnt 0" ::: "memory");
  __builtin_amdgcn_wave_barrier();
}

// A-fragment (16x32 f16): lane(m,half) holds 16 contiguous halfs:
// K = k0 + half*16 + [0,16) (matches B-frag order)
__device__ __forceinline__ v16h load_a_tile(const _Float16* base, int ldm, int k0,
                                            int m, int half) {
  v8h lo = *reinterpret_cast<const v8h*>(base + m * ldm + k0 + half * 16);
  v8h hi = *reinterpret_cast<const v8h*>(base + m * ldm + k0 + half * 16 + 8);
  v16h a;
#pragma unroll
  for (int i = 0; i < 8; i++) { a[i] = lo[i]; a[8 + i] = hi[i]; }
  return a;
}

// Fragment stored pre-swizzled: frag f, lane l -> 16 contiguous halfs
__device__ __forceinline__ v16h load_frag(const _Float16* base, int f, int lane) {
  return *reinterpret_cast<const v16h*>(base + (f * 32 + lane) * 16);
}

// Convert one (W1,W2) pair into fragment-swizzled f16 layout.
// dst: WFRAG*512 halfs. B-frag order: elem i of lane l is
// value [K = k0 + (l>>4)*16 + i][n0 + (l&15)]
__device__ __forceinline__ void convert_weights(const float* __restrict__ W1g,
                                                const float* __restrict__ W2g,
                                                _Float16* dst, int tid) {
  for (int i = tid; i < 4 * 512; i += 128) {
    int f = i >> 9, l = (i >> 4) & 31, e = i & 15;
    int nn = f * 16 + (l & 15);
    int kk = (l >> 4) * 16 + e;
    dst[i] = (kk < 25 && nn < 50) ? (_Float16)W1g[kk * 50 + nn] : (_Float16)0.f;
  }
  for (int i = tid; i < 14 * 512; i += 128) {
    int f = i >> 9, l = (i >> 4) & 31, e = i & 15;
    int nb = f >> 1, kh = f & 1;
    int nn = nb * 16 + (l & 15);
    int kk = kh * 32 + (l >> 4) * 16 + e;
    dst[4 * 512 + i] =
        (kk < 50 && nn < 100) ? (_Float16)W2g[kk * 100 + nn] : (_Float16)0.f;
  }
}

// One block per idx (16 total): build swizzled f16 weights + padded biases in ws
__global__ __launch_bounds__(128) void prep_weights_kernel(
    const float* __restrict__ W1, const float* __restrict__ b1,
    const float* __restrict__ W2, const float* __restrict__ b2,
    _Float16* __restrict__ wsW, float* __restrict__ wsB) {
  const int idx = blockIdx.x;
  const int tid = threadIdx.x;
  convert_weights(W1 + (size_t)idx * 25 * 50, W2 + (size_t)idx * 50 * 100,
                  wsW + (size_t)idx * (WFRAG * 512), tid);
  float* bd = wsB + idx * 176;
  for (int i = tid; i < 64; i += 128)  bd[i]      = (i < 50)  ? b1[idx * 50 + i]  : 0.f;
  for (int i = tid; i < 112; i += 128) bd[64 + i] = (i < 100) ? b2[idx * 100 + i] : 0.f;
}

template <bool PRE>
__global__ __launch_bounds__(128) void deepmd_desc_kernel(
    const float* __restrict__ coords, const int* __restrict__ types,
    const float* __restrict__ W0, const float* __restrict__ b0,
    const float* __restrict__ W1, const float* __restrict__ b1,
    const float* __restrict__ W2, const float* __restrict__ b2,
    const _Float16* __restrict__ wsW, const float* __restrict__ wsB,
    float* __restrict__ out) {
  __shared__ __align__(32) _Float16 sGs[NKT * NNC * 512];   // G, swizzled (100 KB)
  __shared__ __align__(32) _Float16 sT2s[NKT * 512];        // t2^T A-frags (14 KB)
  __shared__ __align__(32) _Float16 sWs[WFRAG * 512];       // W1+W2 B-frags (18 KB)
  __shared__ __align__(32) _Float16 sH0[4][16 * 32];        // per-wave H0 tile
  __shared__ __align__(32) _Float16 sH1[4][16 * 64];        // per-wave H1 tile
  __shared__ float sEnvA[NPAD * 3];
  __shared__ float sEnvR[NPAD];
  __shared__ float sB1[64];
  __shared__ float sB2[112];
  __shared__ float sT[12];                                  // t[c][a]
  __shared__ int   sCnt[4], sOff[5], sCur[4];

  const int atom = blockIdx.x;
  const int bi = atom / NATOM, n = atom % NATOM;
  const int tid = threadIdx.x;
  const int lane = tid & 31, wave = tid >> 5;
  const int m = lane & 15, half = lane >> 4;

  // ---------------- phase 0: zero-init + bucketed local environment --------
  for (int i = tid; i < (NKT * NNC * 512) / 2; i += 128)
    reinterpret_cast<int*>(sGs)[i] = 0;
  for (int i = tid; i < NPAD * 3; i += 128) sEnvA[i] = 0.f;
  for (int i = tid; i < NPAD; i += 128) sEnvR[i] = 0.f;
  if (tid < 12) sT[tid] = 0.f;
  if (tid < 4) sCnt[tid] = 0;
  __syncthreads();

  const int ti = types[n];
  const float cx = coords[(bi * NATOM + n) * 3 + 0];
  const float cy = coords[(bi * NATOM + n) * 3 + 1];
  const float cz = coords[(bi * NATOM + n) * 3 + 2];

  for (int k = tid; k < NNBR; k += 128) {
    int j = (k < n) ? k : k + 1;
    atomicAdd(&sCnt[types[j]], 1);
  }
  __syncthreads();
  if (tid == 0) {
    int o = 0;
    for (int t = 0; t < 4; t++) {
      sOff[t] = o; sCur[t] = o;
      o += (sCnt[t] + 15) & ~15;
    }
    sOff[4] = o;
  }
  __syncthreads();

  for (int k = tid; k < NNBR; k += 128) {
    int j = (k < n) ? k : k + 1;
    int tj = types[j];
    int row = atomicAdd(&sCur[tj], 1);
    float dx = cx - coords[(bi * NATOM + j) * 3 + 0];
    float dy = cy - coords[(bi * NATOM + j) * 3 + 1];
    float dz = cz - coords[(bi * NATOM + j) * 3 + 2];
    float d2 = dx * dx + dy * dy + dz * dz;
    float invd = rsqrtf(d2);
    float invd2 = invd * invd;
    sEnvR[row] = invd;
    sEnvA[row * 3 + 0] = dx * invd2;
    sEnvA[row * 3 + 1] = dy * invd2;
    sEnvA[row * 3 + 2] = dz * invd2;
  }
  __syncthreads();
  const int total_pad = sOff[4];

  // ---------------- phase 1: per-bucket fused 3-layer MLP via WMMA ---------
  for (int t = 0; t < 4; t++) {
    const int idx = ti * 4 + t;
    if (PRE) {
      // async DMA of pre-swizzled f16 weight fragments (ASYNCcnt-tracked)
      const _Float16* wsrc = wsW + (size_t)idx * (WFRAG * 512);
#if HAVE_ASYNC_LDS
      {
        // builtin takes int4-vector pointers (b128 = one int4 per lane)
        __attribute__((address_space(1))) v4i* gp =
            reinterpret_cast<__attribute__((address_space(1))) v4i*>(
                (uintptr_t)wsrc);
        __attribute__((address_space(3))) v4i* lp =
            reinterpret_cast<__attribute__((address_space(3))) v4i*>(
                (uint32_t)(uintptr_t)&sWs[0]);
        for (int ch = tid; ch < (WFRAG * 512 * 2) / 16; ch += 128)
          __builtin_amdgcn_global_load_async_to_lds_b128(gp + ch, lp + ch, 0, 0);
      }
#else
      {
        const int4* src = reinterpret_cast<const int4*>(wsrc);
        int4* dst = reinterpret_cast<int4*>(&sWs[0]);
        for (int ch = tid; ch < (WFRAG * 512 * 2) / 16; ch += 128) dst[ch] = src[ch];
      }
#endif
      for (int i = tid; i < 176; i += 128) {
        float v = wsB[idx * 176 + i];
        if (i < 64) sB1[i] = v; else sB2[i - 64] = v;
      }
#if HAVE_ASYNC_LDS
      wait_async0();
#endif
    } else {
      convert_weights(W1 + (size_t)idx * 25 * 50, W2 + (size_t)idx * 50 * 100,
                      &sWs[0], tid);
      for (int i = tid; i < 64; i += 128)
        sB1[i] = (i < 50) ? b1[idx * 50 + i] : 0.f;
      for (int i = tid; i < 112; i += 128)
        sB2[i] = (i < 100) ? b2[idx * 100 + i] : 0.f;
    }
    __syncthreads();

    const int ntile = (sOff[t + 1] - sOff[t]) >> 4;
    const float* W0g = W0 + idx * 25;
    const float* b0g = b0 + idx * 25;

    for (int tl = wave; tl < ntile; tl += 4) {
      const int row0 = sOff[t] + tl * 16;
      // layer 0: h0 = tanh(env_r * w0 + b0)
      float er = sEnvR[row0 + m];
      for (int c = half; c < 32; c += 2) {
        float v = (c < 25) ? fast_tanh(er * W0g[c] + b0g[c]) : 0.f;
        sH0[wave][m * 32 + c] = (_Float16)v;
      }
      lds_wave_fence();

      // layer 1: H1 = tanh(H0 @ W1 + b1) + [H0,H0]
      v16h a0 = load_a_tile(&sH0[wave][0], 32, 0, m, half);
#pragma unroll
      for (int nb = 0; nb < 4; nb++) {
        v16h bf = load_frag(sWs, nb, lane);
        v8f c = {};
        c = __builtin_amdgcn_wmma_f32_16x16x32_f16(false, a0, false, bf,
                                                   (short)0, c, false, false);
        int col = nb * 16 + m;
#pragma unroll
        for (int r = 0; r < 8; r++) {
          int rr = r + half * 8;
          float v = 0.f;
          if (col < 50)
            v = fast_tanh(c[r] + sB1[col]) + (float)sH0[wave][rr * 32 + (col % 25)];
          sH1[wave][rr * 64 + col] = (_Float16)v;
        }
      }
      lds_wave_fence();

      // layer 2: G = tanh(H1 @ W2 + b2) + [H1,H1]; accumulate t[c][a]
      v16h a1 = load_a_tile(&sH1[wave][0], 64, 0, m, half);
      v16h a2 = load_a_tile(&sH1[wave][0], 64, 32, m, half);
#pragma unroll
      for (int nb = 0; nb < 7; nb++) {
        v16h bfa = load_frag(sWs, 4 + nb * 2 + 0, lane);
        v16h bfb = load_frag(sWs, 4 + nb * 2 + 1, lane);
        v8f c = {};
        c = __builtin_amdgcn_wmma_f32_16x16x32_f16(false, a1, false, bfa,
                                                   (short)0, c, false, false);
        c = __builtin_amdgcn_wmma_f32_16x16x32_f16(false, a2, false, bfb,
                                                   (short)0, c, false, false);
        int col = nb * 16 + m;
#pragma unroll
        for (int r = 0; r < 8; r++) {
          int rr = r + half * 8;
          int rg = row0 + rr;
          float g = 0.f;
          if (col < 100)
            g = fast_tanh(c[r] + sB2[col]) + (float)sH1[wave][rr * 64 + (col % 50)];
          // write G straight into B-fragment order for phase 2:
          // frag (kt = rg>>5, nc = nb), lane = (col&15) + 16*((rg>>4)&1), elem = rg&15
          int fdst = (rg >> 5) * NNC + nb;
          int ldst = (col & 15) + ((rg >> 4) & 1) * 16;
          sGs[(fdst * 32 + ldst) * 16 + (rg & 15)] = (_Float16)g;
          if (col < 4) {
            atomicAdd(&sT[0 * 4 + col], sEnvA[rg * 3 + 0] * g);
            atomicAdd(&sT[1 * 4 + col], sEnvA[rg * 3 + 1] * g);
            atomicAdd(&sT[2 * 4 + col], sEnvA[rg * 3 + 2] * g);
          }
        }
      }
    }
    __syncthreads();
  }

  // ------- phase 2: build t2^T A-fragments, then D = t2^T @ G via WMMA -----
  const int nkt = (total_pad + 31) >> 5;
  // element (kt, l, e): a = l&15, K = kt*32 + (l>>4)*16 + e
  for (int i = tid; i < nkt * 512; i += 128) {
    int kt = i >> 9, l = (i >> 4) & 31, e = i & 15;
    int aa = l & 15, hf = l >> 4;
    int K = kt * 32 + hf * 16 + e;
    float v = 0.f;
    if (aa < 4)
      v = sEnvA[K * 3 + 0] * sT[aa] + sEnvA[K * 3 + 1] * sT[4 + aa] +
          sEnvA[K * 3 + 2] * sT[8 + aa];
    sT2s[i] = (_Float16)v;
  }
  __syncthreads();

  for (int nc = wave; nc < NNC; nc += 4) {
    v8f acc = {};
    for (int kt = 0; kt < nkt; kt++) {
      v16h af = load_frag(sT2s, kt, lane);
      v16h bf = load_frag(sGs, kt * NNC + nc, lane);
      acc = __builtin_amdgcn_wmma_f32_16x16x32_f16(false, af, false, bf,
                                                   (short)0, acc, false, false);
    }
    if (half == 0) {
#pragma unroll
      for (int r = 0; r < 4; r++) {          // row M=r is axis index a
        int mm = nc * 16 + m;
        if (mm < 100)
          out[((size_t)atom * 100 + mm) * 4 + r] = acc[r];
      }
    }
  }
}

extern "C" void kernel_launch(void* const* d_in, const int* in_sizes, int n_in,
                              void* d_out, int out_size, void* d_ws, size_t ws_size,
                              hipStream_t stream) {
  (void)in_sizes; (void)n_in; (void)out_size;
  const float* coords = (const float*)d_in[0];
  const int*   types  = (const int*)d_in[1];
  const float* W0 = (const float*)d_in[2];
  const float* b0 = (const float*)d_in[3];
  const float* W1 = (const float*)d_in[4];
  const float* b1 = (const float*)d_in[5];
  const float* W2 = (const float*)d_in[6];
  const float* b2 = (const float*)d_in[7];
  float* out = (float*)d_out;

  const size_t needW = (size_t)16 * WFRAG * 512 * 2;   // swizzled f16 weights
  const size_t needB = (size_t)16 * 176 * 4;           // padded biases
  if (d_ws && ws_size >= needW + needB) {
    _Float16* wsW = (_Float16*)d_ws;
    float*    wsB = (float*)((char*)d_ws + needW);
    prep_weights_kernel<<<dim3(16), 128, 0, stream>>>(W1, b1, W2, b2, wsW, wsB);
    deepmd_desc_kernel<true><<<dim3(BATCH * NATOM), 128, 0, stream>>>(
        coords, types, W0, b0, W1, b1, W2, b2, wsW, wsB, out);
  } else {
    deepmd_desc_kernel<false><<<dim3(BATCH * NATOM), 128, 0, stream>>>(
        coords, types, W0, b0, W1, b1, W2, b2, nullptr, nullptr, out);
  }
}
